// Net_75806172774760
// MI455X (gfx1250) — matrix operand order repaired and checked by reading, not compiled, not run on hardware
//
#include <hip/hip_runtime.h>
#include <math.h>

// ---------------------------------------------------------------------------
// Problem dims (fixed by the reference): B=256, N=512, F=4, K=256, H=32, C=2
// ---------------------------------------------------------------------------
#define DB 256
#define DN 512
#define DK 256
#define DH 32

typedef __attribute__((ext_vector_type(16))) __bf16 bf16x16;
typedef __attribute__((ext_vector_type(8)))  __bf16 bf16x8;
typedef __attribute__((ext_vector_type(8)))  float  v8f;
typedef __attribute__((ext_vector_type(8)))  float  f32x8;
typedef __attribute__((ext_vector_type(8)))  unsigned short ushort8;

union FragBF {
    bf16x16 bf;
    bf16x8  h8[2];
    ushort8 u8[2];
};

__device__ __forceinline__ unsigned short f2bf(float f) {
    unsigned int u = __float_as_uint(f);
    u += 0x7FFFu + ((u >> 16) & 1u);      // round-to-nearest-even
    return (unsigned short)(u >> 16);
}

__device__ __forceinline__ void store_out(float* p, float v)          { *p = v; }
__device__ __forceinline__ void store_out(unsigned short* p, float v) { *p = f2bf(v); }

// ---------------------------------------------------------------------------
// LDS-free batched GEMM:  C[b] = A[b] * B[b]  (+bias, relu) with BF16 WMMA.
//   A  : row-major [M][K]   (f32 -> converted in-register, or bf16)
//   BT : row-major [N][K]   (bf16, k-contiguous = transposed B)
// Per the wave32 16x16x32 bf16 fragment layouts, each lane's A and B fragments
// are two contiguous 16B chunks -> direct global b128 loads, no LDS, no
// barriers. Lanes l and l+16 consume alternating 16B chunks of one row, so every
// 64B line is fully used. Cross-block operand reuse is served by the 192MB L2
// (per-batch working set ~1.5MB).
//   Block: 128 threads = 4 waves; wave w owns m-stripe (16 rows), NT n-tiles.
//   STORE_T: write C transposed as [N][M] (ldc = row stride of that matrix).
// ---------------------------------------------------------------------------
template <typename TA, bool BIAS_RELU, bool STORE_T, typename TOUT, int NT>
__global__ __launch_bounds__(128)
void gemm_bf16_wmma(const TA* __restrict__ A, long long strideA, int lda,
                    const unsigned short* __restrict__ BT, long long strideB, int ldbT,
                    const float* __restrict__ bias,
                    TOUT* __restrict__ C, long long strideC, int ldc,
                    int Kdim)
{
    constexpr int BN = 16 * NT;

    const int b    = blockIdx.z;
    const int mBlk = blockIdx.y * 64;
    const int nBlk = blockIdx.x * BN;

    const int t    = threadIdx.x;
    const int wave = t >> 5;
    const int lane = t & 31;
    const int lg   = lane >> 4;     // lane group (0: lanes 0-15, 1: lanes 16-31)
    const int lm   = lane & 15;

    const int mrow = mBlk + wave * 16 + lm;
    const TA* __restrict__ Arow = A + (long long)b * strideA + (long long)mrow * lda;
    const unsigned short* __restrict__ Bb = BT + (long long)b * strideB;
    TOUT* __restrict__ Cb = C + (long long)b * strideC;

    // Per-lane B row pointers (n = nBlk + 16*j + lm), k-contiguous
    const unsigned short* __restrict__ Brow[NT];
    #pragma unroll
    for (int j = 0; j < NT; ++j)
        Brow[j] = Bb + (long long)(nBlk + 16 * j + lm) * ldbT;

    v8f acc[NT];
    #pragma unroll
    for (int j = 0; j < NT; ++j) acc[j] = {};

    #pragma unroll 2
    for (int k0 = 0; k0 < Kdim; k0 += 32) {
        // --- A fragment: K = {8g..8g+7, 16+8g..23+8g} at row mrow ---
        FragBF fa;
        if constexpr (sizeof(TA) == 4) {
            f32x8 c0 = *(const f32x8*)&Arow[k0 + 8 * lg];
            f32x8 c1 = *(const f32x8*)&Arow[k0 + 16 + 8 * lg];
            fa.h8[0] = __builtin_convertvector(c0, bf16x8);
            fa.h8[1] = __builtin_convertvector(c1, bf16x8);
        } else {
            fa.u8[0] = *(const ushort8*)&Arow[k0 + 8 * lg];
            fa.u8[1] = *(const ushort8*)&Arow[k0 + 16 + 8 * lg];
        }
        // --- B fragments: K = {8g..8g+7, 16+8g..23+8g} at row n ---
        #pragma unroll
        for (int j = 0; j < NT; ++j) {
            FragBF fb;
            fb.u8[0] = *(const ushort8*)&Brow[j][k0 + 8 * lg];
            fb.u8[1] = *(const ushort8*)&Brow[j][k0 + 16 + 8 * lg];
            acc[j] = __builtin_amdgcn_wmma_f32_16x16x32_bf16(false, fa.bf, false, fb.bf,
                                                             (short)0, acc[j], false, false);
        }
    }

    // --- epilogue: acc[j][r] is (row = mBase + r, col = nBlk + 16j + lm) ---
    const int mBase = mBlk + wave * 16 + 8 * lg;
    #pragma unroll
    for (int j = 0; j < NT; ++j) {
        const int col = nBlk + 16 * j + lm;
        float bia = 0.f;
        if (BIAS_RELU) bia = bias[col];
        #pragma unroll
        for (int r = 0; r < 8; ++r) {
            const int row = mBase + r;
            float v = acc[j][r];
            if (BIAS_RELU) { v += bia; v = v > 0.f ? v : 0.f; }
            if (STORE_T) store_out(&Cb[(long long)col * ldc + row], v);   // 8 contiguous/lane
            else         store_out(&Cb[(long long)row * ldc + col], v);
        }
    }
}

// ---------------------------------------------------------------------------
// xW1T[b,h,n] = sum_f x[b,n,f] * W1[f,h]   (F=4, H=32) -> bf16, transposed
// ---------------------------------------------------------------------------
__global__ __launch_bounds__(128)
void xw1_kernel(const float* __restrict__ x, const float* __restrict__ W1,
                unsigned short* __restrict__ xW1T)
{
    __shared__ float W1s[4 * 32];
    const int t = threadIdx.x;
    W1s[t] = W1[t];                      // 128 == 4*32 exactly
    __syncthreads();

    const long long idx = (long long)blockIdx.x * 128 + t;   // b*N + n
    const long long bb  = idx >> 9;                           // batch
    const int       n   = (int)(idx & 511);
    const float* xp = x + idx * 5;                            // x[...,:4], drop bias col
    const float f0 = xp[0], f1 = xp[1], f2 = xp[2], f3 = xp[3];
    unsigned short* dst = xW1T + bb * (DH * DN) + n;          // [b][h][n]
    #pragma unroll
    for (int h = 0; h < 32; ++h) {
        float acc = f0 * W1s[h] + f1 * W1s[32 + h] + f2 * W1s[64 + h] + f3 * W1s[96 + h];
        dst[(long long)h * DN] = f2bf(acc);
    }
}

// ---------------------------------------------------------------------------
// s = softmax(h @ Ws + bs), fused; emits sT[b][k][n] (transposed, bf16).
// hT input is [b][32][512]. K = H = 32 -> one WMMA k-step. 4 waves x 4 N-tiles.
// ---------------------------------------------------------------------------
__global__ __launch_bounds__(128)
void s_softmax_kernel(const unsigned short* __restrict__ hT,    // [B][32][512]
                      const float* __restrict__ Ws,             // [32][256]
                      const float* __restrict__ bs,             // [256]
                      unsigned short* __restrict__ sT)          // [B][256][512]
{
    __shared__ __align__(16) unsigned short WsT[256][32];  // [n][k]
    __shared__ __align__(16) unsigned short Ah[16][32];    // [m][k]
    __shared__ float spre[16][256];
    __shared__ float rinv[16];

    const int b  = blockIdx.y;
    const int m0 = blockIdx.x * 16;
    const int t  = threadIdx.x;

    for (int i = t; i < 32 * 256; i += 128) {
        const int k = i >> 8, n = i & 255;
        WsT[n][k] = f2bf(Ws[k * 256 + n]);
    }
    for (int i = t; i < 16 * 32; i += 128) {
        const int r = i >> 5, k = i & 31;                       // m=m0+r, k=h
        Ah[r][k] = hT[(long long)b * (DH * DN) + (long long)k * DN + m0 + r];
    }
    __syncthreads();

    const int wave = t >> 5, lane = t & 31, lg = lane >> 4, lm = lane & 15;
    FragBF fa;
    fa.u8[0] = *(const ushort8*)&Ah[lm][8 * lg];
    fa.u8[1] = *(const ushort8*)&Ah[lm][16 + 8 * lg];

    #pragma unroll
    for (int nt = 0; nt < 4; ++nt) {
        const int nTile = wave * 4 + nt;
        FragBF fb;
        fb.u8[0] = *(const ushort8*)&WsT[nTile * 16 + lm][8 * lg];
        fb.u8[1] = *(const ushort8*)&WsT[nTile * 16 + lm][16 + 8 * lg];
        v8f c = {};
        c = __builtin_amdgcn_wmma_f32_16x16x32_bf16(false, fa.bf, false, fb.bf,
                                                    (short)0, c, false, false);
        #pragma unroll
        for (int r = 0; r < 8; ++r)
            spre[r + 8 * lg][nTile * 16 + lm] = c[r];
    }
    __syncthreads();

    if (t < 16) {
        float mx = -3.4e38f;
        for (int j = 0; j < 256; ++j) {
            const float v = spre[t][j] + bs[j];
            spre[t][j] = v;
            mx = v > mx ? v : mx;
        }
        float sum = 0.f;
        for (int j = 0; j < 256; ++j) {
            const float e = __expf(spre[t][j] - mx);
            spre[t][j] = e;
            sum += e;
        }
        rinv[t] = 1.f / sum;
    }
    __syncthreads();

    // cooperative transposed store: sT[b][j][m0+r]
    unsigned short* dstB = sT + (long long)b * (DK * DN) + m0;
    for (int i = t; i < 256 * 16; i += 128) {
        const int j = i >> 4, r = i & 15;
        dstB[(long long)j * DN + r] = f2bf(spre[r][j] * rinv[r]);
    }
}

// ---------------------------------------------------------------------------
// a_pool normalization: zero diag, d = row sums, scale by dinv_i * dinv_j -> bf16
// ---------------------------------------------------------------------------
__global__ __launch_bounds__(256)
void pool_norm_kernel(const float* __restrict__ ap, unsigned short* __restrict__ apbf)
{
    __shared__ float dinv[256];
    const int b = blockIdx.x, i = threadIdx.x;
    const float* row = ap + ((long long)b * DK + i) * DK;
    float s = 0.f;
    for (int j = 0; j < DK; ++j)
        if (j != i) s += row[j];
    dinv[i] = (s > 0.f) ? rsqrtf(s) : 0.f;
    __syncthreads();
    const float di = dinv[i];
    unsigned short* dst = apbf + ((long long)b * DK + i) * DK;
    for (int j = 0; j < DK; ++j) {
        const float v = (j == i) ? 0.f : row[j] * di * dinv[j];
        dst[j] = f2bf(v);
    }
}

// ---------------------------------------------------------------------------
// xW2T[b,h,k] = x_pool[b,k,:] @ W2[:,h]  (32x32) -> bf16, transposed
// ---------------------------------------------------------------------------
__global__ __launch_bounds__(128)
void xw2_kernel(const float* __restrict__ xp, const float* __restrict__ W2,
                unsigned short* __restrict__ outT)
{
    __shared__ float W2s[32 * 32];
    const int t = threadIdx.x;
    for (int i = t; i < 1024; i += 128) W2s[i] = W2[i];
    __syncthreads();

    const long long idx = (long long)blockIdx.x * 128 + t;   // b*K + k
    const long long bb  = idx >> 8;
    const int       k   = (int)(idx & 255);
    const float* v = xp + idx * 32;
    float xr[32];
    #pragma unroll
    for (int j = 0; j < 32; ++j) xr[j] = v[j];
    unsigned short* dst = outT + bb * (DH * DK) + k;          // [b][h][k]
    for (int h = 0; h < 32; ++h) {
        float acc = 0.f;
        #pragma unroll
        for (int j = 0; j < 32; ++j) acc += xr[j] * W2s[j * 32 + h];
        dst[(long long)h * DK] = f2bf(acc);
    }
}

// ---------------------------------------------------------------------------
// out[b,:] = h2[b].sum(axis=0) @ Wd + bd
// ---------------------------------------------------------------------------
__global__ __launch_bounds__(32)
void final_kernel(const float* __restrict__ h2, const float* __restrict__ Wd,
                  const float* __restrict__ bd, float* __restrict__ out)
{
    __shared__ float cs[32];
    const int b = blockIdx.x, h = threadIdx.x;
    const float* base = h2 + (long long)b * DK * DH;
    float s = 0.f;
    for (int k = 0; k < DK; ++k) s += base[k * DH + h];
    cs[h] = s;
    __syncthreads();
    if (h < 2) {
        float acc = bd[h];
        #pragma unroll
        for (int j = 0; j < 32; ++j) acc += cs[j] * Wd[j * 2 + h];
        out[b * 2 + h] = acc;
    }
}

// ---------------------------------------------------------------------------
// Launch pipeline. Workspace layout (bytes), with lifetime-safe aliasing:
//   [0,   8M)   xW1T bf16 [B][32][512]   -> later x_pool f32 [B][256][32]
//   [8M, 16M)   hT   bf16 [B][32][512]   -> later xW2T bf16 [B][32][256]
//   [16M, 80M)  sT   bf16 [B][256][512]  -> later a_pool_norm bf16 [B][256][256]
//   [80M,144M)  asT  bf16 [B][256][512]  -> later h2 f32 [B][256][32]
//   [144M,208M) a_pool f32 [B][256][256]
// Requires ws_size >= 208 MB.
// ---------------------------------------------------------------------------
extern "C" void kernel_launch(void* const* d_in, const int* in_sizes, int n_in,
                              void* d_out, int out_size, void* d_ws, size_t ws_size,
                              hipStream_t stream)
{
    const float* x  = (const float*)d_in[0];
    const float* a  = (const float*)d_in[1];
    const float* W1 = (const float*)d_in[2];
    const float* b1 = (const float*)d_in[3];
    const float* Ws = (const float*)d_in[4];
    const float* bs = (const float*)d_in[5];
    const float* W2 = (const float*)d_in[6];
    const float* b2 = (const float*)d_in[7];
    const float* Wd = (const float*)d_in[8];
    const float* bd = (const float*)d_in[9];
    float* out = (float*)d_out;

    char* ws = (char*)d_ws;
    const size_t MB = 1u << 20;
    unsigned short* xW1T  = (unsigned short*)(ws);
    unsigned short* hT    = (unsigned short*)(ws + 8 * MB);
    unsigned short* sT    = (unsigned short*)(ws + 16 * MB);
    unsigned short* asT   = (unsigned short*)(ws + 80 * MB);
    float*          a_pool = (float*)(ws + 144 * MB);
    // aliases (after last use of the original occupant)
    float*          x_pool   = (float*)xW1T;
    unsigned short* xW2T     = hT;
    unsigned short* apool_bf = sT;
    float*          h2       = (float*)asT;

    // 1) xW1T = (xf @ W1)^T
    xw1_kernel<<<(DB * DN) / 128, 128, 0, stream>>>(x, W1, xW1T);

    // 2) hT = relu(a @ xW1 + b1)^T   [M=512,N=32,K=512], A=f32, STORE_T, NT=2
    gemm_bf16_wmma<float, true, true, unsigned short, 2>
        <<<dim3(1, DN / 64, DB), 128, 0, stream>>>(
            a, (long long)DN * DN, DN,
            xW1T, (long long)DH * DN, DN,
            b1,
            hT, (long long)DH * DN, DN, DN);

    // 3) sT = softmax(h @ Ws + bs)^T  fused
    s_softmax_kernel<<<dim3(DN / 16, DB), 128, 0, stream>>>(hT, Ws, bs, sT);

    // 4) asT = (a @ s)^T             [M=512,N=256,K=512], A=f32, STORE_T, NT=4
    gemm_bf16_wmma<float, false, true, unsigned short, 4>
        <<<dim3(DK / 64, DN / 64, DB), 128, 0, stream>>>(
            a, (long long)DN * DN, DN,
            sT, (long long)DK * DN, DN,
            nullptr,
            asT, (long long)DK * DN, DN, DN);

    // 5) x_pool = s^T @ h            [M=256,N=32,K=512]: A=sT (plain), B=hT, NT=2
    gemm_bf16_wmma<unsigned short, false, false, float, 2>
        <<<dim3(1, DK / 64, DB), 128, 0, stream>>>(
            sT, (long long)DK * DN, DN,
            hT, (long long)DH * DN, DN,
            nullptr,
            x_pool, (long long)DK * DH, DH, DN);

    // 6) a_pool = s^T @ a_s          [M=256,N=256,K=512]: A=sT, B=asT, NT=4
    gemm_bf16_wmma<unsigned short, false, false, float, 4>
        <<<dim3(DK / 64, DK / 64, DB), 128, 0, stream>>>(
            sT, (long long)DK * DN, DN,
            asT, (long long)DK * DN, DN,
            nullptr,
            a_pool, (long long)DK * DK, DK, DN);

    // 7) zero diag + degree normalize -> bf16 (over sT region)
    pool_norm_kernel<<<DB, 256, 0, stream>>>(a_pool, apool_bf);

    // 8) xW2T = (x_pool @ W2)^T -> bf16 (over hT region)
    xw2_kernel<<<(DB * DK) / 128, 128, 0, stream>>>(x_pool, W2, xW2T);

    // 9) h2 = relu(a_pool_norm @ xW2 + b2)  [M=256,N=32,K=256]: A=apool_bf, B=xW2T, NT=2
    gemm_bf16_wmma<unsigned short, true, false, float, 2>
        <<<dim3(1, DK / 64, DB), 128, 0, stream>>>(
            apool_bf, (long long)DK * DK, DK,
            xW2T, (long long)DH * DK, DK,
            b2,
            h2, (long long)DK * DH, DH, DK);

    // 10) out = h2.sum(axis=1) @ Wd + bd
    final_kernel<<<DB, 32, 0, stream>>>(h2, Wd, bd, out);
}
